// MaskedGCN_24524263260253
// MI455X (gfx1250) — matrix operand order repaired and checked by reading, not compiled, not run on hardware
//
#include <hip/hip_runtime.h>
#include <hip/hip_bf16.h>
#include <math.h>

// ---------------------------------------------------------------------------
// Types for CDNA5 WMMA (gfx1250, wave32)
// ---------------------------------------------------------------------------
typedef __attribute__((ext_vector_type(16))) __bf16 v16bf;
typedef __attribute__((ext_vector_type(8)))  __bf16 v8bf;
typedef __attribute__((ext_vector_type(8)))  float  v8f;

__device__ __forceinline__ unsigned short f2bf(float f) {
  unsigned u = __float_as_uint(f);
  unsigned r = u + 0x7FFFu + ((u >> 16) & 1u);   // round-to-nearest-even
  return (unsigned short)(r >> 16);
}

__device__ __forceinline__ void atomAddF32(float* p, float v) {
  unsafeAtomicAdd(p, v);   // hardware global_atomic_add_f32
}

// ---------------------------------------------------------------------------
// Small elementwise helpers
// ---------------------------------------------------------------------------
__global__ void k_set_const(float* __restrict__ p, float v, long long n) {
  long long i = (long long)blockIdx.x * blockDim.x + threadIdx.x;
  if (i < n) p[i] = v;
}

__global__ void k_recip(const float* __restrict__ in, float* __restrict__ out, int n) {
  int i = blockIdx.x * blockDim.x + threadIdx.x;
  if (i < n) out[i] = 1.0f / in[i];
}

// deg accumulation: one thread per edge, atomic on source (row)
__global__ void k_deg_accum(const long long* __restrict__ ei, long long E,
                            float* __restrict__ degf) {
  long long e = (long long)blockIdx.x * blockDim.x + threadIdx.x;
  if (e < E) atomAddF32(&degf[ei[e]], 1.0f);
}

// degf (in: degree incl self-loop) -> degInv in place; dis = rsqrt(deg)
__global__ void k_finish_deg(float* __restrict__ degf, float* __restrict__ dis, long long N) {
  long long i = (long long)blockIdx.x * blockDim.x + threadIdx.x;
  if (i < N) {
    float d = degf[i];
    dis[i]  = rsqrtf(d);
    degf[i] = 1.0f / d;
  }
}

// ---------------------------------------------------------------------------
// Edge kernels: one wave32 per edge, features across lanes (coalesced)
// ---------------------------------------------------------------------------
template <int F>
__global__ void k_edge_mask(const long long* __restrict__ ei, long long E,
                            const float* __restrict__ x,
                            const float* __restrict__ invSig,
                            const float* __restrict__ dis,
                            float* __restrict__ mask) {
  long long wid = (long long)blockIdx.x * (blockDim.x >> 5) + (threadIdx.x >> 5);
  if (wid >= E) return;
  int lane = threadIdx.x & 31;
  long long r = ei[wid];
  long long c = ei[E + wid];
  float ne = dis[r] * dis[c];
  const float* xr = x + r * F;
  const float* xc = x + c * F;
  float* mr = mask + r * F;
#pragma unroll
  for (int f = lane; f < F; f += 32) {
    float d = (xc[f] - xr[f]) * invSig[f];
    atomAddF32(&mr[f], ne * d * d);
  }
}

// agg[col] += dis[r]*dis[c] * xw[row]  (widths 64 / 40, runtime strides)
__global__ void k_edge_agg(const long long* __restrict__ ei, long long E,
                           const float* __restrict__ xw, int ldx,
                           const float* __restrict__ dis,
                           float* __restrict__ agg, int lda, int width) {
  long long wid = (long long)blockIdx.x * (blockDim.x >> 5) + (threadIdx.x >> 5);
  if (wid >= E) return;
  int lane = threadIdx.x & 31;
  long long r = ei[wid];
  long long c = ei[E + wid];
  float ne = dis[r] * dis[c];
  const float* xr = xw + r * (long long)ldx;
  float* ac = agg + c * (long long)lda;
  for (int f = lane; f < width; f += 32)
    atomAddF32(&ac[f], ne * xr[f]);
}

// x_masked (bf16) = x * exp(-mask/deg)
template <int LOG2F>
__global__ void k_apply_mask_bf16(const float* __restrict__ x,
                                  const float* __restrict__ mask,
                                  const float* __restrict__ degInv,
                                  unsigned short* __restrict__ xb, long long n) {
  long long i = (long long)blockIdx.x * blockDim.x + threadIdx.x;
  if (i >= n) return;
  long long node = i >> LOG2F;
  float m = __expf(-mask[i] * degInv[node]);
  xb[i] = f2bf(x[i] * m);
}

// agg init with self-loop term: agg[n][f] = xw[n][f] / deg[n]   (64-wide, stride 64)
__global__ void k_init_agg64(const float* __restrict__ xw, const float* __restrict__ degInv,
                             float* __restrict__ agg, long long n) {
  long long i = (long long)blockIdx.x * blockDim.x + threadIdx.x;
  if (i < n) agg[i] = xw[i] * degInv[i >> 6];
}

// layer-2 variant: xw stride 48, agg stride 40 (real C=40)
__global__ void k_init_agg40(const float* __restrict__ xw, const float* __restrict__ degInv,
                             float* __restrict__ agg, long long N) {
  long long i = (long long)blockIdx.x * blockDim.x + threadIdx.x;
  long long n = i / 40;
  if (n >= N) return;
  int f = (int)(i - n * 40);
  agg[i] = xw[n * 48 + f] * degInv[n];
}

// h = relu(agg + b) in place (stride 64)
__global__ void k_bias_relu64(float* __restrict__ a, const float* __restrict__ b, long long n) {
  long long i = (long long)blockIdx.x * blockDim.x + threadIdx.x;
  if (i < n) a[i] = fmaxf(a[i] + b[i & 63], 0.0f);
}

// ---------------------------------------------------------------------------
// B-operand pre-pack into WMMA lane layout (bf16):
//   Bp[(((kt*NT)+nt)*32 + lane)*8 + v] packs K = kbase+2v, kbase+2v+1 for
//   column N = nt*16 + lane%16, kbase = kt*32 + (lane>=16 ? 8 : 0) + (v>=4 ? 8 : 0).
// ---------------------------------------------------------------------------
template <int K, int NOUT, int NT>
__global__ void k_pack_b(const float* __restrict__ W, unsigned* __restrict__ Bp) {
  const int KT = K / 32;
  int i = blockIdx.x * blockDim.x + threadIdx.x;
  if (i >= KT * NT * 32 * 8) return;
  int v    = i & 7;
  int lane = (i >> 3) & 31;
  int tile = i >> 8;
  int nt   = tile % NT;
  int kt   = tile / NT;
  int n    = nt * 16 + (lane & 15);
  int k0   = kt * 32 + ((lane >= 16) ? 8 : 0) + 2 * v + ((v >= 4) ? 8 : 0);
  float w0 = (n < NOUT) ? W[(long long)k0 * NOUT + n] : 0.0f;
  float w1 = (n < NOUT) ? W[(long long)(k0 + 1) * NOUT + n] : 0.0f;
  Bp[i] = (unsigned)f2bf(w0) | ((unsigned)f2bf(w1) << 16);
}

// ---------------------------------------------------------------------------
// WMMA GEMM: D[N x NT*16] = A(bf16, lda=K) x Bpacked, f32 accumulate.
// One wave per 16-row tile; K-tiles of 32; EXEC stays all-ones (row clamp).
// ---------------------------------------------------------------------------
template <int K, int NT>
__global__ void k_wmma_gemm(const unsigned short* __restrict__ Abf,
                            const unsigned* __restrict__ Bp,
                            float* __restrict__ D, long long N) {
  const int ldD = NT * 16;
  long long wave = ((long long)blockIdx.x * blockDim.x + threadIdx.x) >> 5;
  long long row0 = wave * 16;
  if (row0 >= N) return;                       // wave-uniform exit
  int  lane = threadIdx.x & 31;
  int  m    = lane & 15;
  bool hi   = lane >= 16;
  long long row  = row0 + m;
  long long rowc = (row < N) ? row : (N - 1);  // clamp keeps EXEC full for WMMA

  v8f c[NT];
#pragma unroll
  for (int t = 0; t < NT; ++t) { v8f z = {}; c[t] = z; }

#pragma unroll
  for (int kt = 0; kt < K / 32; ++kt) {
    const unsigned short* ap = Abf + rowc * K + kt * 32 + (hi ? 8 : 0);
    v8bf a0 = *(const v8bf*)(ap);
    v8bf a1 = *(const v8bf*)(ap + 16);
    v16bf av = __builtin_shufflevector(a0, a1, 0,1,2,3,4,5,6,7,8,9,10,11,12,13,14,15);
#pragma unroll
    for (int t = 0; t < NT; ++t) {
      const unsigned* bp = Bp + ((kt * NT + t) * 32 + lane) * 8;
      v16bf bv = *(const v16bf*)(bp);
      c[t] = __builtin_amdgcn_wmma_f32_16x16x32_bf16(
          false, av, false, bv, (short)0, c[t], false, false);
    }
  }

#pragma unroll
  for (int t = 0; t < NT; ++t) {
#pragma unroll
    for (int r = 0; r < 8; ++r) {
      long long rr = row0 + r + (hi ? 8 : 0);
      if (rr < N) D[rr * ldD + t * 16 + m] = c[t][r];
    }
  }
}

// ---------------------------------------------------------------------------
// log_softmax over 40 classes: one wave per node, shfl reductions
// ---------------------------------------------------------------------------
__global__ void k_logsoftmax40(const float* __restrict__ agg, const float* __restrict__ b,
                               float* __restrict__ out, long long N) {
  long long node = ((long long)blockIdx.x * blockDim.x + threadIdx.x) >> 5;
  if (node >= N) return;
  int lane = threadIdx.x & 31;
  const float* a = agg + node * 40;
  float v0 = a[lane] + b[lane];
  float v1 = (lane < 8) ? (a[32 + lane] + b[32 + lane]) : -INFINITY;
  float mx = fmaxf(v0, v1);
#pragma unroll
  for (int o = 16; o; o >>= 1) mx = fmaxf(mx, __shfl_xor(mx, o, 32));
  float s = __expf(v0 - mx) + ((lane < 8) ? __expf(v1 - mx) : 0.0f);
#pragma unroll
  for (int o = 16; o; o >>= 1) s += __shfl_xor(s, o, 32);
  float ls = __logf(s);
  out[node * 40 + lane] = v0 - mx - ls;
  if (lane < 8) out[node * 40 + 32 + lane] = v1 - mx - ls;
}

// ---------------------------------------------------------------------------
// Host-side launch
// ---------------------------------------------------------------------------
static inline unsigned cdiv(long long a, long long b) { return (unsigned)((a + b - 1) / b); }

extern "C" void kernel_launch(void* const* d_in, const int* in_sizes, int n_in,
                              void* d_out, int out_size, void* d_ws, size_t ws_size,
                              hipStream_t stream) {
  const float*     x      = (const float*)d_in[0];
  const long long* ei     = (const long long*)d_in[1];   // int64 [2, E]
  const float*     sigma1 = (const float*)d_in[2];
  const float*     W1     = (const float*)d_in[3];
  const float*     b1     = (const float*)d_in[4];
  const float*     sigma2 = (const float*)d_in[5];
  const float*     W2     = (const float*)d_in[6];
  const float*     b2     = (const float*)d_in[7];
  float*           out    = (float*)d_out;

  const int F = 128, H = 64;                 // fixed by reference
  const long long N = in_sizes[0] / F;
  const long long E = in_sizes[1] / 2;

  // ---- workspace carve (aliased across phases; ~129 MB total) ----
  char*  ws  = (char*)d_ws;
  size_t off = 0;
  auto alloc = [&](size_t bytes) -> char* {
    char* p = ws + off;
    off += (bytes + 255) & ~(size_t)255;
    return p;
  };
  float*          degInv = (float*)alloc(N * 4);            // degree -> 1/deg
  float*          dis    = (float*)alloc(N * 4);            // rsqrt(deg)
  float*          invS1  = (float*)alloc(F * 4);
  float*          invS2  = (float*)alloc(H * 4);
  unsigned*       Bp1    = (unsigned*)alloc(4 * 4 * 32 * 8 * 4);
  unsigned*       Bp2    = (unsigned*)alloc(2 * 3 * 32 * 8 * 4);
  float*          mask1  = (float*)alloc((size_t)N * F * 4);    // reused as mask2
  unsigned short* xb1    = (unsigned short*)alloc((size_t)N * F * 2); // reused as hb2
  float*          xw1    = (float*)alloc((size_t)N * H * 4);    // reused as xw2 (N*48)
  float*          agg1   = (float*)alloc((size_t)N * H * 4);    // -> h1 -> agg2 (N*40)
  (void)n_in; (void)out_size; (void)ws_size;

  float*          mask2 = mask1;
  unsigned short* hb2   = xb1;
  float*          xw2   = xw1;
  float*          agg2  = agg1;

  const int T = 256;

  // ---- norm / degree ----
  k_set_const<<<cdiv(N, T), T, 0, stream>>>(degInv, 1.0f, N);       // self-loop
  k_recip<<<1, F, 0, stream>>>(sigma1, invS1, F);
  k_recip<<<1, H, 0, stream>>>(sigma2, invS2, H);
  k_pack_b<128, 64, 4><<<cdiv(4 * 4 * 32 * 8, T), T, 0, stream>>>(W1, Bp1);
  k_pack_b<64, 40, 3><<<cdiv(2 * 3 * 32 * 8, T), T, 0, stream>>>(W2, Bp2);
  k_deg_accum<<<cdiv(E, T), T, 0, stream>>>(ei, E, degInv);
  k_finish_deg<<<cdiv(N, T), T, 0, stream>>>(degInv, dis, N);

  // ---- layer 1 ----
  k_set_const<<<cdiv(N * F, T), T, 0, stream>>>(mask1, 0.0f, N * F);
  k_edge_mask<128><<<cdiv(E * 32, T), T, 0, stream>>>(ei, E, x, invS1, dis, mask1);
  k_apply_mask_bf16<7><<<cdiv(N * F, T), T, 0, stream>>>(x, mask1, degInv, xb1, N * F);
  k_wmma_gemm<128, 4><<<cdiv(cdiv(N, 16) * 32LL, T), T, 0, stream>>>(xb1, Bp1, xw1, N);
  k_init_agg64<<<cdiv(N * H, T), T, 0, stream>>>(xw1, degInv, agg1, N * H);
  k_edge_agg<<<cdiv(E * 32, T), T, 0, stream>>>(ei, E, xw1, 64, dis, agg1, 64, 64);
  k_bias_relu64<<<cdiv(N * H, T), T, 0, stream>>>(agg1, b1, N * H);   // agg1 -> h1

  // ---- layer 2 ----
  k_set_const<<<cdiv(N * H, T), T, 0, stream>>>(mask2, 0.0f, N * H);
  k_edge_mask<64><<<cdiv(E * 32, T), T, 0, stream>>>(ei, E, agg1, invS2, dis, mask2);
  k_apply_mask_bf16<6><<<cdiv(N * H, T), T, 0, stream>>>(agg1, mask2, degInv, hb2, N * H);
  k_wmma_gemm<64, 3><<<cdiv(cdiv(N, 16) * 32LL, T), T, 0, stream>>>(hb2, Bp2, xw2, N);
  k_init_agg40<<<cdiv(N * 40, T), T, 0, stream>>>(xw2, degInv, agg2, N);
  k_edge_agg<<<cdiv(E * 32, T), T, 0, stream>>>(ei, E, xw2, 48, dis, agg2, 40, 40);

  // ---- output ----
  k_logsoftmax40<<<cdiv(N * 32, T), T, 0, stream>>>(agg2, b2, out, N);
}
